// BeeSender_27281632264214
// MI455X (gfx1250) — compile-verified, matrix operand order checked by reading, not run on hardware
//
#include <hip/hip_runtime.h>
#include <hip/hip_bf16.h>

typedef __bf16 v16bf __attribute__((ext_vector_type(16)));
typedef __bf16 v8bf  __attribute__((ext_vector_type(8)));
typedef __bf16 v4bf  __attribute__((ext_vector_type(4)));
typedef float  v8f   __attribute__((ext_vector_type(8)));

#define RNUM 3
#define FDIM 128
#define HDIM 128
#define EMBD 64
#define KDIM 128   // K is 128 for both layers and the final FC
#define KSTEPS (KDIM / 32)

__device__ __forceinline__ __bf16 f2bf(float x) { return (__bf16)x; }

__device__ __forceinline__ v16bf cat16(v8bf lo, v8bf hi) {
    return __builtin_shufflevector(lo, hi, 0, 1, 2, 3, 4, 5, 6, 7,
                                   8, 9, 10, 11, 12, 13, 14, 15);
}

// ---------------------------------------------------------------- zero fill
__global__ void zero_f32_kernel(float* __restrict__ p, size_t n) {
    size_t i = (size_t)blockIdx.x * blockDim.x + threadIdx.x;
    size_t stride = (size_t)gridDim.x * blockDim.x;
    for (; i < n; i += stride) p[i] = 0.0f;
}

// ---------------------------------------------------------------- edge counts
__global__ void count_kernel(const int* __restrict__ ed, const int* __restrict__ et,
                             float* __restrict__ cnt, int E, int N) {
    size_t i = (size_t)blockIdx.x * blockDim.x + threadIdx.x;
    if (i < (size_t)E)
        atomicAdd(&cnt[(size_t)et[i] * N + ed[i]], 1.0f);
}

// ---------------------------------------------------------------- f32 -> bf16 row-major convert
__global__ void cvt_f32_bf16_kernel(const float* __restrict__ src, __bf16* __restrict__ dst,
                                    size_t n8) {   // n8 = elements/8
    size_t i = (size_t)blockIdx.x * blockDim.x + threadIdx.x;
    if (i >= n8) return;
    float4 a = ((const float4*)src)[i * 2];
    float4 b = ((const float4*)src)[i * 2 + 1];
    v8bf o;
    o[0] = f2bf(a.x); o[1] = f2bf(a.y); o[2] = f2bf(a.z); o[3] = f2bf(a.w);
    o[4] = f2bf(b.x); o[5] = f2bf(b.y); o[6] = f2bf(b.z); o[7] = f2bf(b.w);
    ((v8bf*)dst)[i] = o;
}

// ---------------------------------------------------------------- weight pack
// src: K x Nc row-major f32. dst (bf16): [tn][ks][lane][16] fragment-major so a
// wave's B fragment is one contiguous 32-byte load.
__global__ void pack_weights_kernel(const float* __restrict__ src, __bf16* __restrict__ dst,
                                    int K, int Nc) {
    int o = blockIdx.x * blockDim.x + threadIdx.x;
    if (o >= K * Nc) return;
    int j    = o & 15;
    int lane = (o >> 4) & 31;
    int rem  = o >> 9;                 // tn*(K/32) + ks
    int ksteps = K >> 5;
    int tn = rem / ksteps;
    int ks = rem - tn * ksteps;
    int half = lane >> 4;
    int col  = tn * 16 + (lane & 15);
    int k    = ks * 32 + half * 8 + ((j < 8) ? j : (16 + j - 8));
    dst[o] = f2bf(src[(size_t)k * Nc + col]);
}

// ---------------------------------------------------------------- fused 4-matrix bf16 WMMA GEMM
// A: M x 128 bf16 row-major. Bp: 4 packed weight matrices (fragment-major,
// matStride = K*Nc elems; m=0..2 -> Crel slices, m=3 -> Croot).
// One wave: 2 M-tiles x 1 N-tile x 4 matrices = 32 WMMAs on register-resident B.
__global__ __launch_bounds__(128) void gemm4_bf16_wmma(
        const __bf16* __restrict__ A, const __bf16* __restrict__ Bp,
        float* __restrict__ Crel, float* __restrict__ Croot,
        int M, int Nc, long long crelStride) {
    int wave   = blockIdx.x * (blockDim.x >> 5) + (threadIdx.x >> 5);
    int tilesN = Nc >> 4;
    int tilesM = M >> 4;
    int pairsM = (tilesM + 1) >> 1;
    if (wave >= pairsM * tilesN) return;           // wave-uniform guard
    int tp = wave / tilesN;
    int tn = wave - tp * tilesN;
    int tm0 = tp * 2;

    int lane = threadIdx.x & 31;
    int half = lane >> 4;
    int l15  = lane & 15;

    size_t matStride = (size_t)KDIM * Nc;

    // ---- B fragments: 4 matrices x 4 K-steps, each one 32B contiguous load ----
    v16bf bf[4][KSTEPS];
#pragma unroll
    for (int m = 0; m < 4; ++m)
#pragma unroll
        for (int ks = 0; ks < KSTEPS; ++ks)
            bf[m][ks] = *(const v16bf*)(Bp + m * matStride +
                                        (((size_t)tn * KSTEPS + ks) * 32 + lane) * 16);

    v8f acc[2][4] = {};
#pragma unroll
    for (int t = 0; t < 2; ++t) {
        int tm = tm0 + t;
        if (tm >= tilesM) break;                   // wave-uniform
        const __bf16* arow = A + (size_t)(tm * 16 + l15) * KDIM;
#pragma unroll
        for (int ks = 0; ks < KSTEPS; ++ks) {
            int ka = ks * 32 + half * 8;
            v16bf a = cat16(*(const v8bf*)(arow + ka),
                            *(const v8bf*)(arow + ka + 16));
#pragma unroll
            for (int m = 0; m < 4; ++m)
                acc[t][m] = __builtin_amdgcn_wmma_f32_16x16x32_bf16(
                    false, a, false, bf[m][ks], (short)0, acc[t][m], false, false);
        }
    }

#pragma unroll
    for (int t = 0; t < 2; ++t) {
        int tm = tm0 + t;
        if (tm >= tilesM) break;
#pragma unroll
        for (int m = 0; m < 4; ++m) {
            float* C = (m < 3) ? (Crel + (size_t)m * crelStride) : Croot;
            float* cbase = C + (size_t)(tm * 16 + half * 8) * Nc + tn * 16 + l15;
#pragma unroll
            for (int v = 0; v < 8; ++v)
                cbase[(size_t)v * Nc] = acc[t][m][v];
        }
    }
}

// ---------------------------------------------------------------- edge scatter
// One lane per 4 contiguous features: b128 gather + 4 f32 atomic adds.
__global__ void edge_scatter_kernel(const float* __restrict__ xr, float* __restrict__ agg,
                                    const int* __restrict__ es, const int* __restrict__ ed,
                                    const int* __restrict__ et, int E, int N, int Hd,
                                    int lpeShift) {
    long long gtid = (long long)blockIdx.x * blockDim.x + threadIdx.x;
    long long e = gtid >> lpeShift;
    if (e >= E) return;
    int j4 = (int)(gtid & ((1 << lpeShift) - 1)) * 4;
    int s = es[e], d = ed[e], t = et[e];
    const float4 v = *(const float4*)(xr + ((size_t)t * N + s) * Hd + j4);
    float* dst = agg + ((size_t)t * N + d) * Hd + j4;
    atomicAdd(dst + 0, v.x);
    atomicAdd(dst + 1, v.y);
    atomicAdd(dst + 2, v.z);
    atomicAdd(dst + 3, v.w);
}

// ---------------------------------------------------------------- combine -> bf16
// outbf[n,h] = act( rootf[n,h] + bias[h] + sum_r agg[r,n,h] / max(cnt[r,n],1) )
__global__ void combine_kernel(const float* __restrict__ agg, const float* __restrict__ cnt,
                               const float* __restrict__ rootf, const float* __restrict__ bias,
                               __bf16* __restrict__ outbf, int N, int Hd, int rowShift,
                               int doRelu) {
    size_t q = (size_t)blockIdx.x * blockDim.x + threadIdx.x;   // one quad per thread
    size_t total = ((size_t)N * Hd) >> 2;
    if (q >= total) return;
    int n  = (int)(q >> rowShift);                  // rowShift = log2(Hd/4)
    int h4 = (int)(q & ((1 << rowShift) - 1)) * 4;

    float4 s  = *(const float4*)(rootf + (size_t)n * Hd + h4);
    float4 bv = *(const float4*)(bias + h4);
    s.x += bv.x; s.y += bv.y; s.z += bv.z; s.w += bv.w;
#pragma unroll
    for (int r = 0; r < RNUM; ++r) {
        float inv = 1.0f / fmaxf(cnt[(size_t)r * N + n], 1.0f);
        float4 a = *(const float4*)(agg + ((size_t)r * N + n) * Hd + h4);
        s.x += a.x * inv; s.y += a.y * inv; s.z += a.z * inv; s.w += a.w * inv;
    }
    if (doRelu) {
        s.x = fmaxf(s.x, 0.f); s.y = fmaxf(s.y, 0.f);
        s.z = fmaxf(s.z, 0.f); s.w = fmaxf(s.w, 0.f);
    }
    v4bf o;
    o[0] = f2bf(s.x); o[1] = f2bf(s.y); o[2] = f2bf(s.z); o[3] = f2bf(s.w);
    *(v4bf*)(outbf + (size_t)n * Hd + h4) = o;
}

// ---------------------------------------------------------------- final FC
// out[b,:] = tanh( concat(node[nest[b]], node[food[b]]) @ W + bias )
// node: bf16 rows of 64; Wp: packed bf16 (K=128, Nc=128).
__global__ __launch_bounds__(128) void fc_gather_wmma(
        const __bf16* __restrict__ node,
        const int* __restrict__ nest, const int* __restrict__ food,
        const __bf16* __restrict__ Wp, const float* __restrict__ bias,
        float* __restrict__ out, int Bn) {
    int wave   = blockIdx.x * (blockDim.x >> 5) + (threadIdx.x >> 5);
    int tilesN = HDIM >> 4;                         // 8
    int tilesM = Bn >> 4;
    if (wave >= tilesM * tilesN) return;
    int tm = wave / tilesN;
    int tn = wave - tm * tilesN;

    int lane = threadIdx.x & 31;
    int half = lane >> 4;
    int l15  = lane & 15;

    int bi = tm * 16 + l15;
    const __bf16* srcA = node + (size_t)nest[bi] * EMBD;   // k in [0,64)
    const __bf16* srcB = node + (size_t)food[bi] * EMBD;   // k in [64,128)

    v8f acc = {};
#pragma unroll
    for (int ks = 0; ks < KSTEPS; ++ks) {
        int ka = ks * 32 + half * 8;
        int kb = ka + 16;
        // 8-element groups are 8-aligned, never straddle the 64 boundary
        const __bf16* pa = (ka < EMBD) ? (srcA + ka) : (srcB + ka - EMBD);
        const __bf16* pb = (kb < EMBD) ? (srcA + kb) : (srcB + kb - EMBD);
        v16bf a = cat16(*(const v8bf*)pa, *(const v8bf*)pb);
        v16bf b = *(const v16bf*)(Wp + (((size_t)tn * KSTEPS + ks) * 32 + lane) * 16);
        acc = __builtin_amdgcn_wmma_f32_16x16x32_bf16(false, a, false, b,
                                                      (short)0, acc, false, false);
    }
    int col = tn * 16 + l15;
    float bv = bias[col];
    float* obase = out + (size_t)(tm * 16 + half * 8) * HDIM + col;
#pragma unroll
    for (int v = 0; v < 8; ++v)
        obase[(size_t)v * HDIM] = tanhf(acc[v] + bv);
}

// ---------------------------------------------------------------- launch
extern "C" void kernel_launch(void* const* d_in, const int* in_sizes, int n_in,
                              void* d_out, int out_size, void* d_ws, size_t ws_size,
                              hipStream_t stream) {
    const float* x        = (const float*)d_in[0];
    const int*   edge_src = (const int*)  d_in[1];
    const int*   edge_dst = (const int*)  d_in[2];
    const int*   edge_typ = (const int*)  d_in[3];
    // d_in[4] = edge_attr (dead code in reference)
    const int*   nest     = (const int*)  d_in[5];
    const int*   food     = (const int*)  d_in[6];
    const float* W_rel1   = (const float*)d_in[7];
    const float* W_root1  = (const float*)d_in[8];
    const float* b1       = (const float*)d_in[9];
    const float* W_rel2   = (const float*)d_in[10];
    const float* W_root2  = (const float*)d_in[11];
    const float* b2       = (const float*)d_in[12];
    const float* fc_W     = (const float*)d_in[13];
    const float* fc_b     = (const float*)d_in[14];
    float* out = (float*)d_out;

    const int N  = in_sizes[0] / FDIM;
    const int E  = in_sizes[1];
    const int Bn = in_sizes[5];

    // ---- workspace layout (256B-aligned bump allocator, aggressive reuse) ----
    char*  base = (char*)d_ws;
    size_t off  = 0;
    auto alloc = [&](size_t bytes) -> void* {
        void* p = base + off;
        off += (bytes + 255) & ~(size_t)255;
        return p;
    };
    float*  cnt  = (float*) alloc(sizeof(float) * RNUM * N);
    float*  xr   = (float*) alloc(sizeof(float) * (size_t)RNUM * N * HDIM); // xr1 -> xr2
    float*  agg  = (float*) alloc(sizeof(float) * (size_t)RNUM * N * HDIM); // agg1 -> agg2
    float*  root = (float*) alloc(sizeof(float) * (size_t)N * HDIM);        // root1 -> root2
    __bf16* abf  = (__bf16*)alloc(sizeof(__bf16) * (size_t)N * KDIM);       // xbf -> h1bf -> nodebf
    __bf16* wp1  = (__bf16*)alloc(sizeof(__bf16) * 4 * KDIM * HDIM);
    __bf16* wp2  = (__bf16*)alloc(sizeof(__bf16) * 4 * KDIM * EMBD);
    __bf16* wpfc = (__bf16*)alloc(sizeof(__bf16) * KDIM * HDIM);

    const int ZB = 256;
    auto zgrid = [](size_t n, int tb) { size_t g = (n + tb - 1) / tb; return (unsigned)(g > 65535 ? 65535 : g); };

    // ---- pack all weights to bf16 fragment-major (tiny) ----
    {
        int n1 = KDIM * HDIM, n2 = KDIM * EMBD;
        for (int m = 0; m < RNUM; ++m)
            pack_weights_kernel<<<(n1 + 255) / 256, 256, 0, stream>>>(
                W_rel1 + (size_t)m * n1, wp1 + (size_t)m * n1, KDIM, HDIM);
        pack_weights_kernel<<<(n1 + 255) / 256, 256, 0, stream>>>(W_root1, wp1 + (size_t)3 * n1, KDIM, HDIM);
        for (int m = 0; m < RNUM; ++m)
            pack_weights_kernel<<<(n2 + 255) / 256, 256, 0, stream>>>(
                W_rel2 + (size_t)m * n2, wp2 + (size_t)m * n2, KDIM, EMBD);
        pack_weights_kernel<<<(n2 + 255) / 256, 256, 0, stream>>>(W_root2, wp2 + (size_t)3 * n2, KDIM, EMBD);
        pack_weights_kernel<<<(n1 + 255) / 256, 256, 0, stream>>>(fc_W, wpfc, KDIM, HDIM);
    }

    // ---- convert x to bf16 ----
    {
        size_t n8 = ((size_t)N * FDIM) >> 3;
        cvt_f32_bf16_kernel<<<(unsigned)((n8 + 255) / 256), 256, 0, stream>>>(x, abf, n8);
    }

    // ---- edge counts (shared by both layers) ----
    zero_f32_kernel<<<zgrid((size_t)RNUM * N, ZB), ZB, 0, stream>>>(cnt, (size_t)RNUM * N);
    count_kernel<<<(E + 255) / 256, 256, 0, stream>>>(edge_dst, edge_typ, cnt, E, N);

    // ---- layer 1: xr[r] = x @ W_rel1[r], root = x @ W_root1 (fused, MT=2) ----
    {
        int waves  = (((N >> 4) + 1) >> 1) * (HDIM >> 4);
        int blocks = (waves + 3) / 4;
        gemm4_bf16_wmma<<<blocks, 128, 0, stream>>>(abf, wp1, xr, root,
                                                    N, HDIM, (long long)N * HDIM);
    }

    // ---- layer 1 scatter + combine (h1 -> abf as bf16) ----
    zero_f32_kernel<<<zgrid((size_t)RNUM * N * HDIM, ZB), ZB, 0, stream>>>(agg, (size_t)RNUM * N * HDIM);
    {
        long long thr = (long long)E * (HDIM >> 2);
        edge_scatter_kernel<<<(unsigned)((thr + 255) / 256), 256, 0, stream>>>(
            xr, agg, edge_src, edge_dst, edge_typ, E, N, HDIM, 5);
    }
    {
        size_t quads = ((size_t)N * HDIM) >> 2;
        combine_kernel<<<(unsigned)((quads + 255) / 256), 256, 0, stream>>>(
            agg, cnt, root, b1, abf, N, HDIM, 5, 1);
    }

    // ---- layer 2: xr2[r] = h1 @ W_rel2[r], root2 = h1 @ W_root2 (fused, MT=2) ----
    {
        int waves  = (((N >> 4) + 1) >> 1) * (EMBD >> 4);
        int blocks = (waves + 3) / 4;
        gemm4_bf16_wmma<<<blocks, 128, 0, stream>>>(abf, wp2, xr, root,
                                                    N, EMBD, (long long)N * EMBD);
    }

    // ---- layer 2 scatter + combine (node -> abf as bf16) ----
    zero_f32_kernel<<<zgrid((size_t)RNUM * N * EMBD, ZB), ZB, 0, stream>>>(agg, (size_t)RNUM * N * EMBD);
    {
        long long thr = (long long)E * (EMBD >> 2);
        edge_scatter_kernel<<<(unsigned)((thr + 255) / 256), 256, 0, stream>>>(
            xr, agg, edge_src, edge_dst, edge_typ, E, N, EMBD, 4);
    }
    {
        size_t quads = ((size_t)N * EMBD) >> 2;
        combine_kernel<<<(unsigned)((quads + 255) / 256), 256, 0, stream>>>(
            agg, cnt, root, b2, abf, N, EMBD, 4, 0);
    }

    // ---- final: gather pairs, FC, tanh ----
    {
        int waves  = (Bn >> 4) * (HDIM >> 4);
        int blocks = (waves + 3) / 4;
        fc_gather_wmma<<<blocks, 128, 0, stream>>>(abf, nest, food, wpfc, fc_b, out, Bn);
    }
}
